// fullgraph_encoder_44813688766725
// MI455X (gfx1250) — matrix operand order repaired and checked by reading, not compiled, not run on hardware
//
#include <hip/hip_runtime.h>
#include <hip/hip_fp16.h>
#include <hip/hip_bf16.h>

typedef __attribute__((ext_vector_type(16))) _Float16 v16h;
typedef __attribute__((ext_vector_type(8)))  float    v8f;

#define DIMC   41
#define TIMC   64
#define NKC    128
#define NHC    4
#define DHC    32
#define BATCH  8
#define TSTEPS 100
#define LPAD   2176
#define TDN    (TSTEPS * DIMC)      // 4100
#define MROWS  (BATCH * LPAD)       // 17408
#define NEGV   (-1.0e10f)
#define SCALE  0.08838834764831845f // 1/sqrt(128)

// ---------------------------------------------------------------------------
// WMMA + async helpers (CDNA5)
// ---------------------------------------------------------------------------
static __device__ __forceinline__ v8f wmma16(v16h a, v16h b, v8f c) {
  return __builtin_amdgcn_wmma_f32_16x16x32_f16(false, a, false, b,
                                                (short)0, c, false, false);
}

// 16B global -> LDS async copy (ASYNCcnt-tracked, no VGPR round trip).
static __device__ __forceinline__ void async_copy16(void* lds_dst,
                                                    const void* gsrc) {
  uint32_t off = (uint32_t)(uintptr_t)lds_dst;  // low 32 bits = LDS offset
  asm volatile("global_load_async_to_lds_b128 %0, %1, off"
               :: "v"(off), "v"(gsrc) : "memory");
}
static __device__ __forceinline__ void wait_async0() {
  asm volatile("s_wait_asynccnt 0x0" ::: "memory");
}

// A fragment 16x32 from row-major f16 buffer (stride ld halves).
// Lanes 0-15 hold K in {0..15}, lanes 16-31 hold K in {16..31}.
static __device__ __forceinline__ v16h afrag_lds(const _Float16* s, int row0,
                                                 int k0, int ld) {
  const int lane = threadIdx.x & 31;
  const _Float16* p = s + (row0 + (lane & 15)) * ld + k0 + ((lane < 16) ? 0 : 16);
  v16h a;
#pragma unroll
  for (int j = 0; j < 16; ++j) a[j] = p[j];
  return a;
}

// B fragment 32x16 where B[k][n] = base[(n0+n)*rowStride + k0 + k].
// Per-lane: 16 contiguous halves (32B) -> two b128 loads.
static __device__ __forceinline__ v16h bfrag_rows(const _Float16* base,
                                                  long rowStride, int n0, int k0) {
  const int lane = threadIdx.x & 31;
  const _Float16* p = base + (long)(n0 + (lane & 15)) * rowStride + k0 +
                      ((lane < 16) ? 0 : 16);
  v16h b;
#pragma unroll
  for (int j = 0; j < 16; ++j) b[j] = p[j];
  return b;
}

// ---------------------------------------------------------------------------
// Kernel 1: stable compaction + feature build (x0 in f16, padded 107->128)
// ---------------------------------------------------------------------------
__global__ __launch_bounds__(256)
void compact_kernel(const float* __restrict__ ctx, const float* __restrict__ value,
                    const int* __restrict__ mask, const float* __restrict__ tval,
                    const int* __restrict__ tmask,
                    const float* __restrict__ timeW, const float* __restrict__ timeB,
                    _Float16* __restrict__ xh, float* __restrict__ outTU,
                    float* __restrict__ outTM, int* __restrict__ nv)
{
  const int b = blockIdx.x;
  const int t = threadIdx.x;
  __shared__ int psum[256];
  const int CH = (TDN + 255) / 256;  // 17
  const int base = t * CH;
  int cnt = 0;
  for (int i = 0; i < CH; ++i) {
    int idx = base + i;
    if (idx < TDN && mask[b * TDN + idx]) cnt++;
  }
  psum[t] = cnt;
  __syncthreads();
  for (int off = 1; off < 256; off <<= 1) {  // inclusive Hillis-Steele scan
    int add = (t >= off) ? psum[t - off] : 0;
    __syncthreads();
    psum[t] += add;
    __syncthreads();
  }
  const int total = psum[255];
  int dest = psum[t] - cnt;  // exclusive prefix = stable destination
  if (t == 0) nv[b] = (total < LPAD) ? total : LPAD;

  for (int i = 0; i < CH; ++i) {
    int idx = base + i;
    if (idx < TDN && mask[b * TDN + idx]) {
      if (dest < LPAD) {
        float tc = ctx[b * TSTEPS + idx / DIMC];
        float u  = value[b * TDN + idx];
        float tu = tval[b * TDN + idx];
        float tm = (float)tmask[b * TDN + idx];
        int   c  = idx % DIMC;
        _Float16* row = xh + ((long)(b * LPAD + dest)) * NKC;
        for (int k = 0; k < TIMC; ++k)
          row[k] = (_Float16)sinf(tc * timeW[k] + timeB[k]);
        for (int k = 0; k < DIMC; ++k)
          row[TIMC + k] = (_Float16)((k == c) ? 1.0f : 0.0f);
        row[105] = (_Float16)u;
        row[106] = (_Float16)tm;
        for (int k = 107; k < NKC; ++k) row[k] = (_Float16)0.0f;
        outTU[b * LPAD + dest] = tu;
        outTM[b * LPAD + dest] = tm;
      }
      dest++;
    }
  }
  // Padded rows: T_=0 -> sin(time_b); C_inds_=0 -> one-hot at channel 0.
  for (int j = total + t; j < LPAD; j += 256) {
    _Float16* row = xh + ((long)(b * LPAD + j)) * NKC;
    for (int k = 0; k < TIMC; ++k) row[k] = (_Float16)sinf(timeB[k]);
    row[TIMC] = (_Float16)1.0f;
    for (int k = TIMC + 1; k < NKC; ++k) row[k] = (_Float16)0.0f;
    outTU[b * LPAD + j] = 0.0f;
    outTM[b * LPAD + j] = 0.0f;
  }
}

// ---------------------------------------------------------------------------
// Kernel 2: QKV projection GEMM (WMMA). grid = (MROWS/128, 3), block = 256.
// A-tile staged via async global->LDS; B-fragment register double-buffered.
// ---------------------------------------------------------------------------
#define LDS_K 64
#define LDA   (LDS_K + 8)  // 72 halves -> 144B row stride (16B aligned, no conflicts)

__global__ __launch_bounds__(256)
void gemm_qkv_kernel(const _Float16* __restrict__ xh,
                     const float* __restrict__ Wq, const float* __restrict__ Wk,
                     const float* __restrict__ Wv,
                     const float* __restrict__ bq, const float* __restrict__ bk,
                     const float* __restrict__ bv, int kin,
                     float* __restrict__ qf, _Float16* __restrict__ qh,
                     _Float16* __restrict__ kh, _Float16* __restrict__ vT)
{
  __shared__ _Float16 sA[128 * LDA];
  __shared__ _Float16 sB[128 * LDA];  // weights transposed: sB[n*LDA + k]
  const int mode = blockIdx.y;        // 0=Q 1=K 2=V
  const float* W    = (mode == 0) ? Wq : (mode == 1) ? Wk : Wv;
  const float* bias = (mode == 0) ? bq : (mode == 1) ? bk : bv;
  const int t  = threadIdx.x;
  const int wv = t >> 5;
  const int mbase = blockIdx.x * 128;

  v8f acc[8];
#pragma unroll
  for (int i = 0; i < 8; ++i) acc[i] = (v8f){0, 0, 0, 0, 0, 0, 0, 0};

  for (int s = 0; s < 2; ++s) {  // K staged 64 at a time
    {  // A tile: async copy 128 rows x 64 halves (4x16B per thread)
      int r  = t >> 1;
      int c0 = (t & 1) * 32;
      const _Float16* src = xh + (long)(mbase + r) * NKC + s * 64 + c0;
      _Float16* dst = sA + r * LDA + c0;
#pragma unroll
      for (int i = 0; i < 32; i += 8) async_copy16(dst + i, src + i);
    }
    // Weights: convert f32->f16, store transposed (overlaps with async A copy)
    for (int idx = t; idx < 64 * 128; idx += 256) {
      int kl = idx >> 7;
      int n  = idx & 127;
      int kg = s * 64 + kl;
      float w = (kg < kin) ? W[kg * NKC + n] : 0.0f;
      sB[n * LDA + kl] = (_Float16)w;
    }
    wait_async0();
    __syncthreads();
#pragma unroll
    for (int ks = 0; ks < 2; ++ks) {
      v16h a = afrag_lds(sA, wv * 16, ks * 32, LDA);
      v16h bnext = bfrag_rows(sB, LDA, 0, ks * 32);
#pragma unroll
      for (int nt = 0; nt < 8; ++nt) {
        v16h bcur = bnext;
        if (nt < 7) bnext = bfrag_rows(sB, LDA, (nt + 1) * 16, ks * 32);
        acc[nt] = wmma16(a, bcur, acc[nt]);
      }
    }
    __syncthreads();
  }

  const int lane = t & 31;
  const int roff = (lane < 16) ? 0 : 8;
  const int ncol = lane & 15;
#pragma unroll
  for (int nt = 0; nt < 8; ++nt) {
#pragma unroll
    for (int r = 0; r < 8; ++r) {
      int  n = nt * 16 + ncol;
      long m = mbase + wv * 16 + r + roff;
      float v = acc[nt][r] + bias[n];
      if (mode == 0) {
        qf[m * NKC + n] = v;
        qh[m * NKC + n] = (_Float16)v;
      } else if (mode == 1) {
        kh[m * NKC + n] = (_Float16)v;
      } else {  // V stored transposed: vT[b][h][d][l]
        int h = n >> 5, d = n & 31;
        int bb = (int)(m / LPAD);
        int l  = (int)(m - (long)bb * LPAD);
        vT[(((long)bb * NHC + h) * DHC + d) * LPAD + l] = (_Float16)v;
      }
    }
  }
}

// ---------------------------------------------------------------------------
// Kernel 3: flash-style masked attention + head-space residual.
// grid = (LPAD/16, NHC, BATCH), block = 32 (one wave). 64 keys per iteration.
// ---------------------------------------------------------------------------
#define LDP 72  // sP row stride in halves (144B: 16B-aligned, conflict-free)

__global__ __launch_bounds__(32)
void attn_kernel(const _Float16* __restrict__ qh, const _Float16* __restrict__ kh,
                 const _Float16* __restrict__ vT, const float* __restrict__ qf,
                 float* __restrict__ of, _Float16* __restrict__ oh,
                 const int* __restrict__ nv)
{
  const int qt = blockIdx.x, h = blockIdx.y, b = blockIdx.z;
  const int lane  = threadIdx.x & 31;
  const int nvb   = nv[b];
  const int qbase = qt * 16;
  const long rowbase = (long)b * LPAD;
  const int ncol = lane & 15;
  const int roff = (lane < 16) ? 0 : 8;
  __shared__ _Float16 sP[16 * LDP];

  v16h qa;  // Q fragment: rows = q tile, K = dh (32)
  {
    const _Float16* p = qh + (rowbase + qbase + ncol) * NKC + h * DHC +
                        ((lane < 16) ? 0 : 16);
#pragma unroll
    for (int j = 0; j < 16; ++j) qa[j] = p[j];
  }

  float rM[8], rL[8];
  v8f o0 = (v8f){0, 0, 0, 0, 0, 0, 0, 0};
  v8f o1 = (v8f){0, 0, 0, 0, 0, 0, 0, 0};
#pragma unroll
  for (int r = 0; r < 8; ++r) { rM[r] = -3.0e38f; rL[r] = 0.0f; }

  const _Float16* khead = kh + rowbase * NKC + h * DHC;
  const _Float16* vhead = vT + ((long)(b * NHC + h) * DHC) * LPAD;

  for (int kt = 0; kt < LPAD / 64; ++kt) {  // 34 iterations of 64 keys
    const int key0 = kt * 64;
    if (kt + 1 < LPAD / 64) {
      __builtin_prefetch(khead + (long)(key0 + 64 + ncol) * NKC, 0, 1);
      __builtin_prefetch(vhead + (long)ncol * LPAD + key0 + 64, 0, 1);
    }

    // --- S = Q K^T over 64 keys: 4 WMMAs ---
    v8f s0, s1, s2, s3;
    {
      const _Float16* p0 = khead + (long)(key0 + ncol) * NKC +
                           ((lane < 16) ? 0 : 16);
      v16h bk0, bk1, bk2, bk3;
#pragma unroll
      for (int j = 0; j < 16; ++j) {
        bk0[j] = p0[j];
        bk1[j] = p0[16 * NKC + j];
        bk2[j] = p0[32 * NKC + j];
        bk3[j] = p0[48 * NKC + j];
      }
      const v8f zero = (v8f){0, 0, 0, 0, 0, 0, 0, 0};
      s0 = wmma16(qa, bk0, zero);
      s1 = wmma16(qa, bk1, zero);
      s2 = wmma16(qa, bk2, zero);
      s3 = wmma16(qa, bk3, zero);
    }

    const bool kv0 = (key0 + ncol) < nvb;
    const bool kv1 = (key0 + 16 + ncol) < nvb;
    const bool kv2 = (key0 + 32 + ncol) < nvb;
    const bool kv3 = (key0 + 48 + ncol) < nvb;

    __syncthreads();  // sP reads of previous iteration complete
#pragma unroll
    for (int r = 0; r < 8; ++r) {
      const bool qv = (qbase + r + roff) < nvb;
      float a0 = (qv && kv0) ? s0[r] * SCALE : NEGV;
      float a1 = (qv && kv1) ? s1[r] * SCALE : NEGV;
      float a2 = (qv && kv2) ? s2[r] * SCALE : NEGV;
      float a3 = (qv && kv3) ? s3[r] * SCALE : NEGV;
      float m = fmaxf(fmaxf(a0, a1), fmaxf(a2, a3));
      m = fmaxf(m, __shfl_xor(m, 1));   // row-max across the 16 lanes of this row
      m = fmaxf(m, __shfl_xor(m, 2));
      m = fmaxf(m, __shfl_xor(m, 4));
      m = fmaxf(m, __shfl_xor(m, 8));
      float newM = fmaxf(rM[r], m);
      float e0 = __expf(a0 - newM);
      float e1 = __expf(a1 - newM);
      float e2 = __expf(a2 - newM);
      float e3 = __expf(a3 - newM);
      float srow = (e0 + e1) + (e2 + e3);
      srow += __shfl_xor(srow, 1);
      srow += __shfl_xor(srow, 2);
      srow += __shfl_xor(srow, 4);
      srow += __shfl_xor(srow, 8);
      float f = __expf(rM[r] - newM);
      rL[r] = rL[r] * f + srow;
      o0[r] *= f;
      o1[r] *= f;
      rM[r] = newM;
      _Float16* prow = sP + (r + roff) * LDP + ncol;
      prow[0]  = (_Float16)e0;
      prow[16] = (_Float16)e1;
      prow[32] = (_Float16)e2;
      prow[48] = (_Float16)e3;
    }
    __syncthreads();

    // --- O += P V over 64 keys: 4 WMMAs ---
    v16h pa0 = afrag_lds(sP, 0, 0, LDP);
    v16h pa1 = afrag_lds(sP, 0, 32, LDP);
    v16h bv00, bv01, bv10, bv11;
    {
      const _Float16* pv = vhead + (long)ncol * LPAD + key0 +
                           ((lane < 16) ? 0 : 16);
#pragma unroll
      for (int j = 0; j < 16; ++j) {
        bv00[j] = pv[j];                  // keys  0..31, d 0..15
        bv10[j] = pv[32 + j];             // keys 32..63, d 0..15
        bv01[j] = pv[16L * LPAD + j];     // keys  0..31, d 16..31
        bv11[j] = pv[16L * LPAD + 32 + j];// keys 32..63, d 16..31
      }
    }
    o0 = wmma16(pa0, bv00, o0);
    o0 = wmma16(pa1, bv10, o0);
    o1 = wmma16(pa0, bv01, o1);
    o1 = wmma16(pa1, bv11, o1);
  }

#pragma unroll
  for (int r = 0; r < 8; ++r) {
    long m = rowbase + qbase + r + roff;
    float inv = 1.0f / rL[r];
    int n0 = h * DHC + ncol;
    float v0 = qf[m * NKC + n0]      + o0[r] * inv;  // Oh = Qh + A@Vh
    float v1 = qf[m * NKC + n0 + 16] + o1[r] * inv;
    of[m * NKC + n0]      = v0;
    of[m * NKC + n0 + 16] = v1;
    oh[m * NKC + n0]      = (_Float16)v0;  // f16 copy: async A-source for MLP
    oh[m * NKC + n0 + 16] = (_Float16)v1;
  }
}

// ---------------------------------------------------------------------------
// Kernel 4: O = (O + relu(O@Wo + bo)) * valid ; also emits next-layer x (f16)
// ---------------------------------------------------------------------------
__global__ __launch_bounds__(256)
void gemm_mlp_kernel(float* __restrict__ of, const _Float16* __restrict__ oh,
                     const float* __restrict__ Wo, const float* __restrict__ bo,
                     const int* __restrict__ nv, _Float16* __restrict__ xh)
{
  __shared__ _Float16 sA[128 * LDA];
  __shared__ _Float16 sB[128 * LDA];
  const int t  = threadIdx.x;
  const int wv = t >> 5;
  const int mbase = blockIdx.x * 128;

  v8f acc[8];
#pragma unroll
  for (int i = 0; i < 8; ++i) acc[i] = (v8f){0, 0, 0, 0, 0, 0, 0, 0};

  for (int s = 0; s < 2; ++s) {
    {  // A tile from f16 copy of O via async global->LDS
      int r  = t >> 1;
      int c0 = (t & 1) * 32;
      const _Float16* src = oh + (long)(mbase + r) * NKC + s * 64 + c0;
      _Float16* dst = sA + r * LDA + c0;
#pragma unroll
      for (int i = 0; i < 32; i += 8) async_copy16(dst + i, src + i);
    }
    for (int idx = t; idx < 64 * 128; idx += 256) {
      int kl = idx >> 7;
      int n  = idx & 127;
      sB[n * LDA + kl] = (_Float16)Wo[(s * 64 + kl) * NKC + n];
    }
    wait_async0();
    __syncthreads();
#pragma unroll
    for (int ks = 0; ks < 2; ++ks) {
      v16h a = afrag_lds(sA, wv * 16, ks * 32, LDA);
      v16h bnext = bfrag_rows(sB, LDA, 0, ks * 32);
#pragma unroll
      for (int nt = 0; nt < 8; ++nt) {
        v16h bcur = bnext;
        if (nt < 7) bnext = bfrag_rows(sB, LDA, (nt + 1) * 16, ks * 32);
        acc[nt] = wmma16(a, bcur, acc[nt]);
      }
    }
    __syncthreads();
  }

  const int lane = t & 31;
  const int roff = (lane < 16) ? 0 : 8;
  const int ncol = lane & 15;
#pragma unroll
  for (int nt = 0; nt < 8; ++nt) {
#pragma unroll
    for (int r = 0; r < 8; ++r) {
      int  n = nt * 16 + ncol;
      long m = mbase + wv * 16 + r + roff;
      int bb = (int)(m / LPAD);
      int l  = (int)(m - (long)bb * LPAD);
      float val = acc[nt][r] + bo[n];
      val = of[m * NKC + n] + fmaxf(val, 0.0f);
      if (l >= nv[bb]) val = 0.0f;  // x = O * valid
      of[m * NKC + n] = val;
      xh[m * NKC + n] = (_Float16)val;
    }
  }
}

// ---------------------------------------------------------------------------
// Kernel 5: out = x @ out_W + out_b
// ---------------------------------------------------------------------------
__global__ __launch_bounds__(256)
void out_kernel(const float* __restrict__ of, const float* __restrict__ outW,
                const float* __restrict__ outB, const int* __restrict__ nv,
                float* __restrict__ out)
{
  int m = blockIdx.x * 256 + threadIdx.x;
  if (m >= MROWS) return;
  int b = m / LPAD;
  int l = m - b * LPAD;
  float acc = outB[0];
  if (l < nv[b]) {
    const float* row = of + (long)m * NKC;
#pragma unroll 8
    for (int k = 0; k < NKC; ++k) acc += row[k] * outW[k];
  }
  out[m] = acc;
}

// ---------------------------------------------------------------------------
// Host launcher. Param order assumes JAX pytree (alphabetical) flattening:
// [0..4] tensors, [5]=L, [6+8i]={Wk,Wo,Wq,Wv,bk,bo,bq,bv} per layer,
// [30]=out_W [31]=out_b [32]=time_W [33]=time_b
// ---------------------------------------------------------------------------
extern "C" void kernel_launch(void* const* d_in, const int* in_sizes, int n_in,
                              void* d_out, int out_size, void* d_ws, size_t ws_size,
                              hipStream_t stream) {
  (void)in_sizes; (void)n_in; (void)out_size; (void)ws_size;
  const float* ctx   = (const float*)d_in[0];
  const float* value = (const float*)d_in[1];
  const int*   mask  = (const int*)d_in[2];
  const float* tval  = (const float*)d_in[3];
  const int*   tmask = (const int*)d_in[4];
  const float* outW  = (const float*)d_in[30];
  const float* outB  = (const float*)d_in[31];
  const float* timeW = (const float*)d_in[32];
  const float* timeB = (const float*)d_in[33];

  char* w = (char*)d_ws;
  auto take = [&](size_t bytes) -> void* {
    void* p = (void*)w;
    w += (bytes + 255) & ~(size_t)255;
    return p;
  };
  int*      nv = (int*)take(sizeof(int) * BATCH);
  _Float16* xh = (_Float16*)take((size_t)MROWS * NKC * 2);
  _Float16* qh = (_Float16*)take((size_t)MROWS * NKC * 2);
  _Float16* kh = (_Float16*)take((size_t)MROWS * NKC * 2);
  _Float16* vT = (_Float16*)take((size_t)MROWS * NKC * 2);
  _Float16* oh = (_Float16*)take((size_t)MROWS * NKC * 2);
  float*    qf = (float*)take((size_t)MROWS * NKC * 4);
  float*    of = (float*)take((size_t)MROWS * NKC * 4);

  float* out   = (float*)d_out;
  float* outTU = out + MROWS;
  float* outTM = out + 2 * MROWS;

  compact_kernel<<<BATCH, 256, 0, stream>>>(ctx, value, mask, tval, tmask,
                                            timeW, timeB, xh, outTU, outTM, nv);
  for (int lay = 0; lay < 3; ++lay) {
    const float* Wk_ = (const float*)d_in[6 + lay * 8 + 0];
    const float* Wo_ = (const float*)d_in[6 + lay * 8 + 1];
    const float* Wq_ = (const float*)d_in[6 + lay * 8 + 2];
    const float* Wv_ = (const float*)d_in[6 + lay * 8 + 3];
    const float* bk_ = (const float*)d_in[6 + lay * 8 + 4];
    const float* bo_ = (const float*)d_in[6 + lay * 8 + 5];
    const float* bq_ = (const float*)d_in[6 + lay * 8 + 6];
    const float* bv_ = (const float*)d_in[6 + lay * 8 + 7];
    const int kin = (lay == 0) ? 107 : NKC;
    gemm_qkv_kernel<<<dim3(MROWS / 128, 3), 256, 0, stream>>>(
        xh, Wq_, Wk_, Wv_, bq_, bk_, bv_, kin, qf, qh, kh, vT);
    attn_kernel<<<dim3(LPAD / 16, NHC, BATCH), 32, 0, stream>>>(qh, kh, vT, qf,
                                                                of, oh, nv);
    gemm_mlp_kernel<<<MROWS / 128, 256, 0, stream>>>(of, oh, Wo_, bo_, nv, xh);
  }
  out_kernel<<<(MROWS + 255) / 256, 256, 0, stream>>>(of, outW, outB, nv, out);
}
